// CRFModel_67448166416747
// MI455X (gfx1250) — compile-verified, matrix operand order checked by reading, not run on hardware
//
#include <hip/hip_runtime.h>
#include <hip/hip_bf16.h>

// ---------------------------------------------------------------------------
// Shapes: B=32, H=W=256 -> h1 (32,128,128,32) bf16 NHWC
//         conv2 implicit GEMM: M=32*128*128, K=288(=9*32), N=64 via WMMA bf16
//         h2 (32,128,128,64) bf16 NHWC -> pool+emissions -> viterbi
// ---------------------------------------------------------------------------

typedef __attribute__((ext_vector_type(16))) __bf16 v16bf;
typedef __attribute__((ext_vector_type(8)))  float  v8f;

union ABFrag { uint4 u[2]; v16bf v; };

// exact pointee type expected by __builtin_amdgcn_global_load_async_to_lds_b128
typedef int v4i __attribute__((vector_size(16)));
typedef __attribute__((address_space(1))) v4i* gv4i_p;   // __device__ v4i*
typedef __attribute__((address_space(3))) v4i* lv4i_p;   // __shared__ v4i*

#define EPS 1e-5f

// ------------------------- kernel 1: conv1 + bn + relu + pool ---------------
__global__ void __launch_bounds__(256)
conv1_bn_pool(const float* __restrict__ x, const float* __restrict__ w,
              const float* __restrict__ cb, const float* __restrict__ g,
              const float* __restrict__ bb, const float* __restrict__ m,
              const float* __restrict__ v, __bf16* __restrict__ h1) {
  int idx = blockIdx.x * blockDim.x + threadIdx.x;   // b*16384 + py*128 + px
  if (idx >= 32 * 128 * 128) return;
  int px = idx & 127, py = (idx >> 7) & 127, b = idx >> 14;

  // 4x4 input window (top-left at 2*py-1, 2*px-1), zero-padded SAME
  float in[4][4];
  const float* xb = x + (size_t)b * 256 * 256;
  int y0 = 2 * py - 1, x0 = 2 * px - 1;
#pragma unroll
  for (int iy = 0; iy < 4; ++iy) {
    int yy = y0 + iy;
#pragma unroll
    for (int ix = 0; ix < 4; ++ix) {
      int xx = x0 + ix;
      in[iy][ix] = (yy >= 0 && yy < 256 && xx >= 0 && xx < 256) ? xb[yy * 256 + xx] : 0.f;
    }
  }

  __bf16* out = h1 + (size_t)idx * 32;
  for (int c = 0; c < 32; ++c) {
    float wc[9];
#pragma unroll
    for (int k = 0; k < 9; ++k) wc[k] = w[c * 9 + k];
    float scale = g[c] * rsqrtf(v[c] + EPS);
    float bias  = bb[c] + (cb[c] - m[c]) * scale;   // fold conv bias into BN
    float mx = -3.4e38f;
#pragma unroll
    for (int ys = 0; ys < 2; ++ys)
#pragma unroll
      for (int xs = 0; xs < 2; ++xs) {
        float s = 0.f;
#pragma unroll
        for (int ky = 0; ky < 3; ++ky)
#pragma unroll
          for (int kx = 0; kx < 3; ++kx)
            s = fmaf(in[ys + ky][xs + kx], wc[ky * 3 + kx], s);
        mx = fmaxf(mx, fmaf(s, scale, bias));
      }
    out[c] = (__bf16)fmaxf(mx, 0.f);   // relu(max(bn)) == max(relu(bn))
  }
}

// ------------------------- kernel 2: pre-swizzle conv2 weights --------------
// B-fragment image: [chunk(9)][ntile(4)][lane(32)][half(16)], lane -> N column,
// half h -> K = (h&7) + 8*(lane>>4) + 16*(h>>3)   (ISA 16-bit operand layout)
__global__ void pack_w2(const float* __restrict__ w2, __bf16* __restrict__ bp) {
  int idx = blockIdx.x * blockDim.x + threadIdx.x;
  if (idx >= 9 * 4 * 32 * 16) return;
  int h = idx & 15, l = (idx >> 4) & 31, nt = (idx >> 9) & 3, chunk = idx >> 11;
  int ky = chunk / 3, kx = chunk % 3;
  int ic = (h & 7) + 8 * (l >> 4) + 16 * (h >> 3);
  int oc = nt * 16 + (l & 15);
  bp[idx] = (__bf16)w2[((oc * 32 + ic) * 3 + ky) * 3 + kx];
}

// ------------------------- kernel 3: conv2 implicit GEMM via WMMA -----------
__global__ void __launch_bounds__(256)
conv2_wmma(const __bf16* __restrict__ h1, const __bf16* __restrict__ Bp,
           const float* __restrict__ c2b, const float* __restrict__ g,
           const float* __restrict__ bb, const float* __restrict__ m,
           const float* __restrict__ v, __bf16* __restrict__ h2) {
  __shared__ __bf16 Bs[9 * 4 * 32 * 16];   // 36,864 B weight-fragment image
  __shared__ __bf16 otile[8][16 * 64];     // 16 KB epilogue staging

  int tid  = threadIdx.x;
  int lane = tid & 31;
  int wave = tid >> 5;

  // ---- async-stage the whole B image Global -> LDS, once per block --------
  {
    char* gsrc = (char*)Bp;
#pragma unroll
    for (int k = 0; k < 9; ++k) {
      unsigned byte = (unsigned)(k * 256 + tid) * 16u;   // 2304 x 16B = 36,864B
#if __has_builtin(__builtin_amdgcn_global_load_async_to_lds_b128)
      __builtin_amdgcn_global_load_async_to_lds_b128(
          (gv4i_p)(uintptr_t)(gsrc + byte),
          (lv4i_p)(unsigned)(uintptr_t)((char*)&Bs[0] + byte),
          0, 0);
#else
      unsigned loff   = (unsigned)(uintptr_t)((char*)&Bs[0] + byte);
      unsigned long long gaddr = (unsigned long long)(uintptr_t)(gsrc + byte);
      asm volatile("global_load_async_to_lds_b128 %0, %1, off"
                   :: "v"(loff), "v"(gaddr) : "memory");
#endif
    }
#if __has_builtin(__builtin_amdgcn_s_wait_asynccnt)
    __builtin_amdgcn_s_wait_asynccnt(0);
#else
    asm volatile("s_wait_asynccnt 0x0" ::: "memory");
#endif
  }
  __syncthreads();

  int tile = blockIdx.x * 8 + wave;          // 32768 tiles of 16 pixels
  int x0 = (tile & 7) * 16;
  int y  = (tile >> 3) & 127;
  int b  = tile >> 10;
  int hi = lane >> 4;
  int xm = x0 + (lane & 15);                 // this lane's A-row pixel x

  v8f acc[4] = {};
  const __bf16* h1b = h1 + (size_t)b * 128 * 128 * 32;

#pragma unroll
  for (int ky = 0; ky < 3; ++ky) {
    int yy = y + ky - 1;
    bool yok = (yy >= 0) & (yy < 128);
#pragma unroll
    for (int kx = 0; kx < 3; ++kx) {
      int xx = xm + kx - 1;
      bool ok = yok & (xx >= 0) & (xx < 128);
      ABFrag a;
      if (ok) {
        // 32 contiguous bf16 channels at neighbor pixel; this lane needs
        // ch [8*hi,8*hi+8) (halves 0..7) and [16+8*hi,..+8) (halves 8..15)
        const uint4* p = (const uint4*)(h1b + ((size_t)yy * 128 + xx) * 32);
        a.u[0] = p[hi];
        a.u[1] = p[hi + 2];
      } else {
        a.u[0] = make_uint4(0, 0, 0, 0);
        a.u[1] = make_uint4(0, 0, 0, 0);
      }
      int chunk = ky * 3 + kx;
#pragma unroll
      for (int nt = 0; nt < 4; ++nt) {
        ABFrag bm;
        const uint4* q = (const uint4*)&Bs[((chunk * 4 + nt) * 32 + lane) * 16];
        bm.u[0] = q[0];
        bm.u[1] = q[1];
        acc[nt] = __builtin_amdgcn_wmma_f32_16x16x32_bf16(
            false, a.v, false, bm.v, (short)0, acc[nt], false, false);
      }
    }
  }

  // Epilogue: bias + BN + ReLU -> bf16 tile in LDS -> coalesced store
  __bf16* myl = &otile[wave][0];
#pragma unroll
  for (int nt = 0; nt < 4; ++nt) {
    int oc = nt * 16 + (lane & 15);
    float scale = g[oc] * rsqrtf(v[oc] + EPS);
    float bias  = bb[oc] + (c2b[oc] - m[oc]) * scale;
#pragma unroll
    for (int r = 0; r < 8; ++r) {
      int pix = r + hi * 8;                  // D layout: M = r + 8*(lane>>4)
      float val = fmaxf(fmaf(acc[nt][r], scale, bias), 0.f);
      myl[pix * 64 + oc] = (__bf16)val;
    }
  }
  asm volatile("s_wait_dscnt 0" ::: "memory");
  // 16 pixels * 64 oc * 2B = 2048B contiguous in NHWC; 64B per lane
  const uint4* src = (const uint4*)myl + lane * 4;
  uint4* dst = (uint4*)(h2 + (((size_t)b * 128 + y) * 128 + x0) * 64) + lane * 4;
  dst[0] = src[0]; dst[1] = src[1]; dst[2] = src[2]; dst[3] = src[3];
}

// ------------------------- kernel 4: 2x2 maxpool + emission GEMV ------------
__global__ void __launch_bounds__(256)
pool_emissions(const __bf16* __restrict__ h2, const float* __restrict__ emw,
               const float* __restrict__ emb, float* __restrict__ em_out) {
  int idx = blockIdx.x * blockDim.x + threadIdx.x;   // b*4096 + py*64 + px
  if (idx >= 32 * 64 * 64) return;
  int px = idx & 63, py = (idx >> 6) & 63, b = idx >> 12;
  const __bf16* base = h2 + (size_t)b * 128 * 128 * 64;
  const __bf16* p00 = base + ((size_t)(2 * py) * 128 + 2 * px) * 64;
  const __bf16* p01 = p00 + 64;
  const __bf16* p10 = p00 + 128 * 64;
  const __bf16* p11 = p10 + 64;
  float e0 = emb[0], e1 = emb[1], e2 = emb[2], e3 = emb[3];
#pragma unroll 8
  for (int c = 0; c < 64; ++c) {
    float f = fmaxf(fmaxf((float)p00[c], (float)p01[c]),
                    fmaxf((float)p10[c], (float)p11[c]));
    e0 = fmaf(f, emw[c * 4 + 0], e0);
    e1 = fmaf(f, emw[c * 4 + 1], e1);
    e2 = fmaf(f, emw[c * 4 + 2], e2);
    e3 = fmaf(f, emw[c * 4 + 3], e3);
  }
  float* o = em_out + (size_t)idx * 4;
  o[0] = e0; o[1] = e1; o[2] = e2; o[3] = e3;
}

// ------------------------- kernel 5: viterbi + backtrace + head -------------
__global__ void __launch_bounds__(32)
viterbi(const float* __restrict__ em, const float* __restrict__ st,
        const float* __restrict__ et, const float* __restrict__ tr,
        const float* __restrict__ fcw, const float* __restrict__ fcb,
        float* __restrict__ out) {
  __shared__ unsigned char bps[4096 * 4];   // backpointers, LDS-resident
  int b = blockIdx.x;
  int lane = threadIdx.x;
  const float* eb = em + (size_t)b * 4096 * 4;
  if (lane < 4) {
    int c = lane;
    float t0 = tr[0 * 4 + c], t1 = tr[1 * 4 + c];
    float t2 = tr[2 * 4 + c], t3 = tr[3 * 4 + c];
    float score = st[c] + eb[c];
    for (int t = 1; t < 4096; ++t) {
      float s0 = __shfl(score, 0, 4);
      float s1 = __shfl(score, 1, 4);
      float s2 = __shfl(score, 2, 4);
      float s3 = __shfl(score, 3, 4);
      float best = s0 + t0; int arg = 0;
      float cand = s1 + t1; if (cand > best) { best = cand; arg = 1; }
      cand = s2 + t2;       if (cand > best) { best = cand; arg = 2; }
      cand = s3 + t3;       if (cand > best) { best = cand; arg = 3; }
      score = best + eb[t * 4 + c];
      bps[t * 4 + c] = (unsigned char)arg;
    }
    score += et[c];
    float f0 = __shfl(score, 0, 4);
    float f1 = __shfl(score, 1, 4);
    float f2 = __shfl(score, 2, 4);
    float f3 = __shfl(score, 3, 4);
    asm volatile("s_wait_dscnt 0" ::: "memory");
    if (lane == 0) {
      int tag = 0; float bf = f0;
      if (f1 > bf) { bf = f1; tag = 1; }
      if (f2 > bf) { bf = f2; tag = 2; }
      if (f3 > bf) { bf = f3; tag = 3; }
      int counts[4] = {0, 0, 0, 0};
      for (int t = 4095; t >= 1; --t) {   // path[t] = tag, then step back
        counts[tag]++;
        tag = bps[t * 4 + tag];
      }
      counts[tag]++;                       // path[0]
      const float inv = 1.0f / 4096.0f;
#pragma unroll
      for (int n = 0; n < 4; ++n) {
        float s = fcb[n];
#pragma unroll
        for (int c2 = 0; c2 < 4; ++c2)
          s = fmaf((float)counts[c2] * inv, fcw[c2 * 4 + n], s);
        out[b * 4 + n] = s;
      }
    }
  }
}

// ---------------------------------------------------------------------------
extern "C" void kernel_launch(void* const* d_in, const int* in_sizes, int n_in,
                              void* d_out, int out_size, void* d_ws, size_t ws_size,
                              hipStream_t stream) {
  const float* x       = (const float*)d_in[0];
  const float* conv1_w = (const float*)d_in[1];
  const float* conv1_b = (const float*)d_in[2];
  const float* bn1_g   = (const float*)d_in[3];
  const float* bn1_b   = (const float*)d_in[4];
  const float* bn1_m   = (const float*)d_in[5];
  const float* bn1_v   = (const float*)d_in[6];
  const float* conv2_w = (const float*)d_in[7];
  const float* conv2_b = (const float*)d_in[8];
  const float* bn2_g   = (const float*)d_in[9];
  const float* bn2_b   = (const float*)d_in[10];
  const float* bn2_m   = (const float*)d_in[11];
  const float* bn2_v   = (const float*)d_in[12];
  const float* em_w    = (const float*)d_in[13];
  const float* em_b    = (const float*)d_in[14];
  const float* start_t = (const float*)d_in[15];
  const float* end_t   = (const float*)d_in[16];
  const float* trans   = (const float*)d_in[17];
  const float* fc_w    = (const float*)d_in[18];
  const float* fc_b    = (const float*)d_in[19];

  char* ws = (char*)d_ws;
  __bf16* h1 = (__bf16*)(ws);                       // 33,554,432 B
  __bf16* Bp = (__bf16*)(ws + 33554432);            //     36,864 B
  __bf16* h2 = (__bf16*)(ws + 33591296);            // 67,108,864 B

  float* out = (float*)d_out;       // [0,128): final_logits
  float* em  = out + 128;           // [128, 128+524288): emissions

  conv1_bn_pool<<<(32 * 128 * 128 + 255) / 256, 256, 0, stream>>>(
      x, conv1_w, conv1_b, bn1_g, bn1_b, bn1_m, bn1_v, h1);
  pack_w2<<<(9 * 4 * 32 * 16 + 255) / 256, 256, 0, stream>>>(conv2_w, Bp);
  conv2_wmma<<<4096, 256, 0, stream>>>(
      h1, Bp, conv2_b, bn2_g, bn2_b, bn2_m, bn2_v, h2);
  pool_emissions<<<(32 * 64 * 64 + 255) / 256, 256, 0, stream>>>(
      h2, em_w, em_b, em);
  viterbi<<<32, 32, 0, stream>>>(em, start_t, end_t, trans, fc_w, fc_b, out);
}
